// PrismModel_83476984365151
// MI455X (gfx1250) — matrix operand order repaired and checked by reading, not compiled, not run on hardware
//
#include <hip/hip_runtime.h>

// ---------------------------------------------------------------------------
// Types for CDNA5 WMMA (wave32): V_WMMA_F32_16X16X32_BF16
// ---------------------------------------------------------------------------
typedef __bf16 bf16;
typedef bf16  bf16x8   __attribute__((ext_vector_type(8)));
typedef bf16  bf16x16  __attribute__((ext_vector_type(16)));
typedef float floatx4  __attribute__((ext_vector_type(4)));
typedef float floatx8  __attribute__((ext_vector_type(8)));

union BF16x16 { bf16x16 v; bf16x8 h[2]; };

// A-matrix fragment (16x32 bf16): lane L (L<16 -> M=L, K = kb+0..7 & kb+16..23;
// L>=16 -> M=L-16, K = kb+8..15 & kb+24..31). Caller passes p = row_base + kb + lgrp*8.
static __device__ __forceinline__ bf16x16 ld_a_frag(const bf16* p) {
    BF16x16 r;
    r.h[0] = *reinterpret_cast<const bf16x8*>(p);
    r.h[1] = *reinterpret_cast<const bf16x8*>(p + 16);
    return r.v;
}

// B-matrix fragment (32x16 bf16): lane L holds column N=(L&15), K = kb+(L>>4)*8+0..7
// & kb+16+(L>>4)*8+0..7. Caller passes p = col_row_base + kb + lgrp*8.
static __device__ __forceinline__ bf16x16 ld_b_frag(const bf16* p) {
    BF16x16 r;
    r.h[0] = *reinterpret_cast<const bf16x8*>(p);
    r.h[1] = *reinterpret_cast<const bf16x8*>(p + 16);
    return r.v;
}

static __device__ __forceinline__ floatx8 wmma_bf16(bf16x16 a, bf16x16 b, floatx8 c) {
    return __builtin_amdgcn_wmma_f32_16x16x32_bf16(false, a, false, b, (short)0, c,
                                                   false, false);
}

// --- weight staging, split for software pipelining --------------------------
// fetch 32 fp32 into registers (global loads issue; wait sinks past compute)
static __device__ __forceinline__ void fetch32(const float* __restrict__ src,
                                               floatx4 f[8]) {
#pragma unroll
    for (int i = 0; i < 8; ++i)
        f[i] = *reinterpret_cast<const floatx4*>(src + i * 4);
}
// convert fp32 -> bf16 and commit 32 values to LDS (16B vector stores)
static __device__ __forceinline__ void commit32(const floatx4 f[8],
                                                bf16* __restrict__ dst) {
#pragma unroll
    for (int g = 0; g < 8; g += 2) {
        bf16x8 o;
        o[0] = (bf16)f[g].x;     o[1] = (bf16)f[g].y;
        o[2] = (bf16)f[g].z;     o[3] = (bf16)f[g].w;
        o[4] = (bf16)f[g + 1].x; o[5] = (bf16)f[g + 1].y;
        o[6] = (bf16)f[g + 1].z; o[7] = (bf16)f[g + 1].w;
        *reinterpret_cast<bf16x8*>(dst + g * 4) = o;
    }
}
// fused variant for the un-pipelined first chunk
static __device__ __forceinline__ void stage_row32(const float* __restrict__ src,
                                                   bf16* __restrict__ dst) {
    floatx4 f[8];
    fetch32(src, f);
    commit32(f, dst);
}

// ---------------------------------------------------------------------------
// Encoder: x[64-row tile,19] -> 8-layer MLP -> att/rad. grid = (8 tiles, 2 nets)
// Wave tile 32(M) x 64(N); waves arranged 2(M) x 4(N) over the 64x256 block.
// LDS: wchunk0/1 (2 x 16KB) | hA (32KB) | hB (32KB) | x (5KB)
// ---------------------------------------------------------------------------
__global__ void encoder_kernel(
    const float* __restrict__ pos, const float* __restrict__ ue,
    const float* __restrict__ bsa, const float* __restrict__ extra,
    const float* __restrict__ att_w0, const float* __restrict__ att_b0,
    const float* __restrict__ att_wr, const float* __restrict__ att_br,
    const float* __restrict__ rad_w0, const float* __restrict__ rad_b0,
    const float* __restrict__ rad_wr, const float* __restrict__ rad_br,
    float* __restrict__ out_att, float* __restrict__ out_rad,
    bf16* __restrict__ c_ws)
{
    extern __shared__ char smem[];
    bf16*  wb[2];
    wb[0]      = (bf16*)(smem);                  // 256*32 bf16
    wb[1]      = (bf16*)(smem + 16384);          // 256*32 bf16
    bf16*  hA  = (bf16*)(smem + 32768);          // 64*256 bf16
    bf16*  hB  = (bf16*)(smem + 32768 + 32768);  // 64*256 bf16
    float* xb  = (float*)(smem + 32768 + 65536); // 64*20  f32

    const int mt    = blockIdx.x;        // 0..7, 64 rows each
    const int net   = blockIdx.y;        // 0 = att, 1 = rad
    const int mbase = mt * 64;
    const int tid   = threadIdx.x;
    const int w     = tid >> 5;
    const int lane  = tid & 31;
    const int lgrp  = lane >> 4;
    const int lid   = lane & 15;
    const int wm    = w >> 2;            // 0..1 -> M offset wm*32
    const int wn    = w & 3;             // 0..3 -> N offset wn*64

    const float* w0 = net ? rad_w0 : att_w0;
    const float* b0 = net ? rad_b0 : att_b0;
    const float* wr = net ? rad_wr : att_wr;
    const float* br = net ? rad_br : att_br;
    float*     outp = net ? out_rad : out_att;

    // Build x tile [64][19] (stride 20)
    for (int i = tid; i < 64 * 19; i += 256) {
        int r = i / 19, k = i % 19;
        int b = mbase + r;
        float v;
        if      (k < 3) v = pos[b * 3 + k];
        else if (k < 5) v = ue[b * 2 + (k - 3)];
        else if (k < 9) v = bsa[b * 4 + (k - 5)];
        else            v = extra[b * 10 + (k - 9)];
        xb[r * 20 + k] = v;
    }
    __syncthreads();

    // Layer 0: 19 -> 256, scalar (tiny). thread t: row = t&63, cols q*64..q*64+63
    {
        int r = tid & 63, q = tid >> 6;
        for (int n = q * 64; n < q * 64 + 64; ++n) {
            float acc = b0[n];
            const float* wrow = w0 + n * 19;
#pragma unroll
            for (int k = 0; k < 19; ++k) acc += xb[r * 20 + k] * wrow[k];
            hA[r * 256 + n] = (bf16)fmaxf(acc, 0.0f);
        }
    }
    __syncthreads();

    // Layers 1..7: 256 -> 256 via bf16 WMMA, double-buffered weight staging
    bf16* hi = hA;
    bf16* ho = hB;
    floatx8 zero = {};
    for (int layer = 0; layer < 7; ++layer) {
        const float* W  = wr + (size_t)layer * 256 * 256;
        const float* Bb = br + layer * 256;
        floatx8 acc[2][4];
#pragma unroll
        for (int mi = 0; mi < 2; ++mi)
#pragma unroll
            for (int j = 0; j < 4; ++j) acc[mi][j] = zero;

        stage_row32(W + (size_t)tid * 256, wb[0] + tid * 32);
        __syncthreads();

        for (int kc = 0; kc < 8; ++kc) {
            const int kb = kc * 32;
            bf16* cur = wb[kc & 1];
            bf16* nxt = wb[(kc & 1) ^ 1];
            floatx4 pf[8];
            const bool have_next = (kc + 1) < 8;
            if (have_next)  // issue next chunk's global loads now
                fetch32(W + (size_t)tid * 256 + (kc + 1) * 32, pf);

            bf16x16 a0 = ld_a_frag(hi + (wm * 32 + lid) * 256 + kb + lgrp * 8);
            bf16x16 a1 = ld_a_frag(hi + (wm * 32 + 16 + lid) * 256 + kb + lgrp * 8);
#pragma unroll
            for (int j = 0; j < 4; ++j) {
                int n = wn * 64 + j * 16 + lid;
                bf16x16 bfr = ld_b_frag(cur + n * 32 + lgrp * 8);
                acc[0][j] = wmma_bf16(a0, bfr, acc[0][j]);
                acc[1][j] = wmma_bf16(a1, bfr, acc[1][j]);
            }
            if (have_next)  // convert + commit after compute (loadcnt wait lands here)
                commit32(pf, nxt + tid * 32);
            __syncthreads();
        }
        // bias + relu -> ho
#pragma unroll
        for (int j = 0; j < 4; ++j) {
            int n = wn * 64 + j * 16 + lid;
            float bias = Bb[n];
#pragma unroll
            for (int mi = 0; mi < 2; ++mi)
#pragma unroll
                for (int i = 0; i < 8; ++i) {
                    int m = wm * 32 + mi * 16 + i + lgrp * 8;
                    ho[m * 256 + n] = (bf16)fmaxf(acc[mi][j][i] + bias, 0.0f);
                }
        }
        __syncthreads();
        bf16* t = hi; hi = ho; ho = t;
    }

    // Write fp32 att/rad to d_out, bf16 half of c to workspace
    for (int i = tid; i < 64 * 256; i += 256) {
        int r = i >> 8, n = i & 255;
        bf16 hv = hi[i];
        outp[(size_t)(mbase + r) * 256 + n] = (float)hv;
        c_ws[(size_t)(mbase + r) * 512 + net * 256 + n] = hv;
    }
}

// ---------------------------------------------------------------------------
// RF Prism: per-subcarrier 3-layer MLP. grid = (S=1024, 8 B-tiles of 64).
// Wave tile 32(M) x 64(N). Weight chunks double-buffered fp32->bf16 in LDS.
// LDS: wchunk0/1 (2 x 16KB) | h1/h2 (32KB, aliased) | partial (1KB)
// ---------------------------------------------------------------------------
__global__ void prism_kernel(
    const bf16* __restrict__ c_ws,   // [512][512] bf16
    const float* __restrict__ p_w1,  // [1024][256][512]
    const float* __restrict__ p_b1,  // [1024][256]
    const float* __restrict__ p_w2,  // [1024][256][256]
    const float* __restrict__ p_b2,  // [1024][256]
    const float* __restrict__ p_w3,  // [1024][256]
    const float* __restrict__ p_b3,  // [1024]
    float* __restrict__ out_sub)     // [512][1024]
{
    extern __shared__ char smem[];
    bf16* wb[2];
    wb[0]       = (bf16*)(smem);            // 256*32 bf16
    wb[1]       = (bf16*)(smem + 16384);    // 256*32 bf16
    bf16*  h1   = (bf16*)(smem + 32768);    // 64*256 bf16 (h2 aliases this)
    float* pred = (float*)(smem + 32768 + 32768); // 256 f32 partials

    const int s     = blockIdx.x;
    const int mt    = blockIdx.y;
    const int mbase = mt * 64;
    const int tid   = threadIdx.x;
    const int w     = tid >> 5;
    const int lane  = tid & 31;
    const int lgrp  = lane >> 4;
    const int lid   = lane & 15;
    const int wm    = w >> 2;   // 0..1
    const int wn    = w & 3;    // 0..3

    const float* W1 = p_w1 + (size_t)s * 256 * 512;
    const float* W2 = p_w2 + (size_t)s * 256 * 256;

    floatx8 zero = {};
    floatx8 acc[2][4];

    // ---------------- Layer 1: K = 512 (16 chunks) ----------------
#pragma unroll
    for (int mi = 0; mi < 2; ++mi)
#pragma unroll
        for (int j = 0; j < 4; ++j) acc[mi][j] = zero;

    const bf16* arow0 = c_ws + (size_t)(mbase + wm * 32 + lid) * 512;
    const bf16* arow1 = c_ws + (size_t)(mbase + wm * 32 + 16 + lid) * 512;

    stage_row32(W1 + (size_t)tid * 512, wb[0] + tid * 32);
    __syncthreads();
    for (int kc = 0; kc < 16; ++kc) {
        const int kb = kc * 32;
        bf16* cur = wb[kc & 1];
        bf16* nxt = wb[(kc & 1) ^ 1];
        floatx4 pf[8];
        const bool have_next = (kc + 1) < 16;
        if (have_next)
            fetch32(W1 + (size_t)tid * 512 + (kc + 1) * 32, pf);

        bf16x16 a0 = ld_a_frag(arow0 + kb + lgrp * 8);
        bf16x16 a1 = ld_a_frag(arow1 + kb + lgrp * 8);
#pragma unroll
        for (int j = 0; j < 4; ++j) {
            int n = wn * 64 + j * 16 + lid;
            bf16x16 bfr = ld_b_frag(cur + n * 32 + lgrp * 8);
            acc[0][j] = wmma_bf16(a0, bfr, acc[0][j]);
            acc[1][j] = wmma_bf16(a1, bfr, acc[1][j]);
        }
        if (have_next)
            commit32(pf, nxt + tid * 32);
        __syncthreads();
    }
#pragma unroll
    for (int j = 0; j < 4; ++j) {
        int n = wn * 64 + j * 16 + lid;
        float bias = p_b1[s * 256 + n];
#pragma unroll
        for (int mi = 0; mi < 2; ++mi)
#pragma unroll
            for (int i = 0; i < 8; ++i) {
                int m = wm * 32 + mi * 16 + i + lgrp * 8;
                h1[m * 256 + n] = (bf16)fmaxf(acc[mi][j][i] + bias, 0.0f);
            }
    }
    __syncthreads();

    // ---------------- Layer 2: K = 256 (8 chunks) ----------------
#pragma unroll
    for (int mi = 0; mi < 2; ++mi)
#pragma unroll
        for (int j = 0; j < 4; ++j) acc[mi][j] = zero;

    stage_row32(W2 + (size_t)tid * 256, wb[0] + tid * 32);
    __syncthreads();
    for (int kc = 0; kc < 8; ++kc) {
        const int kb = kc * 32;
        bf16* cur = wb[kc & 1];
        bf16* nxt = wb[(kc & 1) ^ 1];
        floatx4 pf[8];
        const bool have_next = (kc + 1) < 8;
        if (have_next)
            fetch32(W2 + (size_t)tid * 256 + (kc + 1) * 32, pf);

        bf16x16 a0 = ld_a_frag(h1 + (wm * 32 + lid) * 256 + kb + lgrp * 8);
        bf16x16 a1 = ld_a_frag(h1 + (wm * 32 + 16 + lid) * 256 + kb + lgrp * 8);
#pragma unroll
        for (int j = 0; j < 4; ++j) {
            int n = wn * 64 + j * 16 + lid;
            bf16x16 bfr = ld_b_frag(cur + n * 32 + lgrp * 8);
            acc[0][j] = wmma_bf16(a0, bfr, acc[0][j]);
            acc[1][j] = wmma_bf16(a1, bfr, acc[1][j]);
        }
        if (have_next)
            commit32(pf, nxt + tid * 32);
        __syncthreads();
    }
    // bias + relu -> h2 (aliases h1; all h1 reads completed at last barrier)
#pragma unroll
    for (int j = 0; j < 4; ++j) {
        int n = wn * 64 + j * 16 + lid;
        float bias = p_b2[s * 256 + n];
#pragma unroll
        for (int mi = 0; mi < 2; ++mi)
#pragma unroll
            for (int i = 0; i < 8; ++i) {
                int m = wm * 32 + mi * 16 + i + lgrp * 8;
                h1[m * 256 + n] = (bf16)fmaxf(acc[mi][j][i] + bias, 0.0f);
            }
    }
    __syncthreads();

    // ---------------- Layer 3: sub[b,s] = h2 . w3[s] + b3[s] ----------------
    {
        int m = tid & 63, q = tid >> 6;
        const float* w3 = p_w3 + s * 256;
        float a3 = 0.0f;
        for (int n = q * 64; n < q * 64 + 64; ++n)
            a3 += (float)h1[m * 256 + n] * w3[n];
        pred[q * 64 + m] = a3;
    }
    __syncthreads();
    if (tid < 64) {
        float r = pred[tid] + pred[64 + tid] + pred[128 + tid] + pred[192 + tid]
                + p_b3[s];
        out_sub[(size_t)(mbase + tid) * 1024 + s] = r;
    }
}

// ---------------------------------------------------------------------------
// MIMO head: [512,1024] x [1024,8]^T + b  (tiny); 4 independent partials for ILP
// ---------------------------------------------------------------------------
__global__ void mimo_kernel(const float* __restrict__ sub,
                            const float* __restrict__ mw,
                            const float* __restrict__ mb,
                            float* __restrict__ out_mimo)
{
    int gid = blockIdx.x * blockDim.x + threadIdx.x;   // 4096 = 512*8
    int b = gid >> 3, o = gid & 7;
    const float* srow = sub + (size_t)b * 1024;
    const float* wrow = mw + o * 1024;
    float p0 = 0.f, p1 = 0.f, p2 = 0.f, p3 = 0.f;
    for (int s4 = 0; s4 < 1024; s4 += 16) {
        floatx4 a0 = *reinterpret_cast<const floatx4*>(srow + s4);
        floatx4 w0 = *reinterpret_cast<const floatx4*>(wrow + s4);
        floatx4 a1 = *reinterpret_cast<const floatx4*>(srow + s4 + 4);
        floatx4 w1 = *reinterpret_cast<const floatx4*>(wrow + s4 + 4);
        floatx4 a2 = *reinterpret_cast<const floatx4*>(srow + s4 + 8);
        floatx4 w2 = *reinterpret_cast<const floatx4*>(wrow + s4 + 8);
        floatx4 a3 = *reinterpret_cast<const floatx4*>(srow + s4 + 12);
        floatx4 w3 = *reinterpret_cast<const floatx4*>(wrow + s4 + 12);
        p0 += a0.x * w0.x + a0.y * w0.y + a0.z * w0.z + a0.w * w0.w;
        p1 += a1.x * w1.x + a1.y * w1.y + a1.z * w1.z + a1.w * w1.w;
        p2 += a2.x * w2.x + a2.y * w2.y + a2.z * w2.z + a2.w * w2.w;
        p3 += a3.x * w3.x + a3.y * w3.y + a3.z * w3.z + a3.w * w3.w;
    }
    out_mimo[b * 8 + o] = ((p0 + p1) + (p2 + p3)) + mb[o];
}

// ---------------------------------------------------------------------------
extern "C" void kernel_launch(void* const* d_in, const int* in_sizes, int n_in,
                              void* d_out, int out_size, void* d_ws, size_t ws_size,
                              hipStream_t stream) {
    (void)in_sizes; (void)n_in; (void)out_size; (void)ws_size;
    const float* pos    = (const float*)d_in[0];
    const float* ue     = (const float*)d_in[1];
    const float* bsa    = (const float*)d_in[2];
    const float* extra  = (const float*)d_in[3];
    const float* att_w0 = (const float*)d_in[4];
    const float* att_b0 = (const float*)d_in[5];
    const float* att_wr = (const float*)d_in[6];
    const float* att_br = (const float*)d_in[7];
    const float* rad_w0 = (const float*)d_in[8];
    const float* rad_b0 = (const float*)d_in[9];
    const float* rad_wr = (const float*)d_in[10];
    const float* rad_br = (const float*)d_in[11];
    const float* p_w1   = (const float*)d_in[12];
    const float* p_b1   = (const float*)d_in[13];
    const float* p_w2   = (const float*)d_in[14];
    const float* p_b2   = (const float*)d_in[15];
    const float* p_w3   = (const float*)d_in[16];
    const float* p_b3   = (const float*)d_in[17];
    const float* mimo_w = (const float*)d_in[18];
    const float* mimo_b = (const float*)d_in[19];

    float* out      = (float*)d_out;
    float* out_sub  = out;                       // 512*1024
    float* out_mimo = out + 512 * 1024;          // 512*8
    float* out_att  = out_mimo + 512 * 8;        // 512*256
    float* out_rad  = out_att + 512 * 256;       // 512*256
    bf16*  c_ws     = (bf16*)d_ws;               // 512*512 bf16 = 512KB

    // Encoder: 16 blocks, ~101KB dynamic LDS each
    size_t smemE = 2 * 16384 + 2 * 32768 + 64 * 20 * 4;
    encoder_kernel<<<dim3(8, 2), 256, smemE, stream>>>(
        pos, ue, bsa, extra,
        att_w0, att_b0, att_wr, att_br,
        rad_w0, rad_b0, rad_wr, rad_br,
        out_att, out_rad, c_ws);

    // Prism: 8192 blocks, ~66KB dynamic LDS each
    size_t smemP = 2 * 16384 + 32768 + 256 * 4;
    prism_kernel<<<dim3(1024, 8), 256, smemP, stream>>>(
        c_ws, p_w1, p_b1, p_w2, p_b2, p_w3, p_b3, out_sub);

    // MIMO head
    mimo_kernel<<<16, 256, 0, stream>>>(out_sub, mimo_w, mimo_b, out_mimo);
}
